// TrajectoryInference_28698971472559
// MI455X (gfx1250) — matrix operand order repaired and checked by reading, not compiled, not run on hardware
//
#include <hip/hip_runtime.h>
#include <hip/hip_bf16.h>
#include <math.h>

// ---------------- problem constants ----------------
#define BSZ   4
#define NPTS  2048
#define DIM   256
#define NCL   50          // NC
#define ROWS  (BSZ * NPTS)
#define SUBSPACE 64       // padded eigen-subspace (need NC+1 = 51)
#define SUB_ITERS 6

typedef __attribute__((ext_vector_type(16))) __bf16 v16bf;
typedef __attribute__((ext_vector_type(8)))  float  v8f;

// float -> bf16, round-to-nearest-even
__device__ __forceinline__ __bf16 f2bf(float f) {
  unsigned int u = __builtin_bit_cast(unsigned int, f);
  unsigned int r = (u + 0x7FFFu + ((u >> 16) & 1u)) >> 16;
  return __builtin_bit_cast(__bf16, (unsigned short)r);
}

// ---- CDNA5 async global->LDS copy (ASYNCcnt path, 16B per lane) ----
__device__ __forceinline__ void async_copy_b128(unsigned lds_off, const void* gaddr) {
  asm volatile("global_load_async_to_lds_b128 %0, %1, off"
               :: "v"(lds_off), "v"(gaddr) : "memory");
}
__device__ __forceinline__ void wait_async0() {
  asm volatile("s_wait_asynccnt 0x0" ::: "memory");
}
__device__ __forceinline__ unsigned lds_addr_of(const void* p) {
  return (unsigned)(unsigned long long)p;   // flat LDS address: low 32 bits
}

// ---------------- block reductions ----------------
__device__ __forceinline__ float block_red_sum(float v, float* sh) {
  const int tid = threadIdx.x;
  sh[tid] = v; __syncthreads();
  for (int s = (int)blockDim.x >> 1; s > 0; s >>= 1) {
    if (tid < s) sh[tid] += sh[tid + s];
    __syncthreads();
  }
  float r = sh[0]; __syncthreads();
  return r;
}
__device__ __forceinline__ float block_red_min(float v, float* sh) {
  const int tid = threadIdx.x;
  sh[tid] = v; __syncthreads();
  for (int s = (int)blockDim.x >> 1; s > 0; s >>= 1) {
    if (tid < s) sh[tid] = fminf(sh[tid], sh[tid + s]);
    __syncthreads();
  }
  float r = sh[0]; __syncthreads();
  return r;
}
__device__ __forceinline__ float block_red_max(float v, float* sh) {
  const int tid = threadIdx.x;
  sh[tid] = v; __syncthreads();
  for (int s = (int)blockDim.x >> 1; s > 0; s >>= 1) {
    if (tid < s) sh[tid] = fmaxf(sh[tid], sh[tid + s]);
    __syncthreads();
  }
  float r = sh[0]; __syncthreads();
  return r;
}

// =====================================================================
// Kernel 1: GEMM + bias (+ exact GELU).  C[M x Npad] = act(A[M x K] @ W[K x Nw] + b)
// One wave = one 16x16 tile via v_wmma_f32_16x16x32_bf16.
// The 32-row W k-slab (contiguous rows) is staged into LDS with
// global_load_async_to_lds_b128, then each lane reads its B fragment from LDS.
// =====================================================================
__global__ void mlp_layer(const float* __restrict__ A, const float* __restrict__ W,
                          const float* __restrict__ bias, float* __restrict__ C,
                          int M, int K, int Nw, int Npad, int gelu_flag) {
  __shared__ float Wsh[32 * 128];                 // up to 16 KB slab
  const int lane  = threadIdx.x;                  // 0..31 (wave32)
  const int tid   = threadIdx.y * 32 + threadIdx.x;
  const int tilesN = Npad >> 4;
  const int tile  = blockIdx.x * blockDim.y + threadIdx.y;   // grids are exact
  const int tm = (tile / tilesN) << 4;
  const int tn = (tile % tilesN) << 4;

  const int m    = tm + (lane & 15);              // A row for this lane
  const int kbA  = (lane >> 4) << 4;              // 0 or 16
  const int ncol = tn + (lane & 15);              // B/C column for this lane
  const int kbB  = (lane >> 4) << 4;              // 0 or 16

  const int   nc  = (ncol < Nw) ? ncol : (Nw - 1);   // clamped (no divergent load)
  const float msk = (ncol < Nw) ? 1.0f : 0.0f;

  const int slab = 32 * Nw * 4;                   // bytes, multiple of 16
  const unsigned lds0 = lds_addr_of(&Wsh[0]);

  v8f acc = {};
  for (int k0 = 0; k0 < K; k0 += 32) {
    // ---- async stage W[k0..k0+31][:] (contiguous) into LDS ----
    const char* gsrc = (const char*)(W + (size_t)k0 * Nw);
    for (int ofs = tid * 16; ofs < slab; ofs += 4096)
      async_copy_b128(lds0 + (unsigned)ofs, gsrc + ofs);
    wait_async0();
    __syncthreads();

    const float* Arow = A + (size_t)m * K + k0;
    __builtin_prefetch((const void*)(Arow + 32), 0, 3);
    v16bf af, bfr;
#pragma unroll
    for (int j = 0; j < 8; ++j)  af[j]     = f2bf(Arow[kbA + j]);
#pragma unroll
    for (int j = 0; j < 8; ++j)  af[8 + j] = f2bf(Arow[kbA + 8 + j]);
#pragma unroll
    for (int j = 0; j < 16; ++j)
      bfr[j] = f2bf(Wsh[(kbB + j) * Nw + nc] * msk);
    acc = __builtin_amdgcn_wmma_f32_16x16x32_bf16(false, af, false, bfr,
                                                  (short)0, acc, false, false);
    __syncthreads();                              // slab reuse barrier
  }
  const float bv = bias[nc] * msk;
  const int rb = tm + ((lane >> 4) << 3);
#pragma unroll
  for (int r = 0; r < 8; ++r) {
    float v = acc[r] + bv;
    if (gelu_flag) v = 0.5f * v * (1.0f + erff(v * 0.70710678118f));
    v *= msk;                                     // keep pad columns zero
    C[(size_t)(rb + r) * Npad + ncol] = v;
  }
}

// =====================================================================
// Kernel 2: squared row norms of X (8192 x 64, cols >= 50 are zero)
// =====================================================================
__global__ void row_sqnorm(const float* __restrict__ X, float* __restrict__ sq, int rows) {
  int r = blockIdx.x * blockDim.x + threadIdx.x;
  if (r >= rows) return;
  const float* x = X + (size_t)r * SUBSPACE;
  float s = 0.f;
#pragma unroll
  for (int j = 0; j < SUBSPACE; ++j) s += x[j] * x[j];
  sq[r] = s;
}

// =====================================================================
// Kernel 3: d2[b][i][j] = max(sq_i + sq_j - 2 x_i.x_j, 0) ; dot via WMMA (K=64).
// Both operands load contiguous rows of X, so no LDS staging needed.
// =====================================================================
__global__ void gram_d2(const float* __restrict__ X, const float* __restrict__ sq,
                        float* __restrict__ d2) {
  const int lane = threadIdx.x;
  const int b    = blockIdx.y;
  const int tile = blockIdx.x * blockDim.y + threadIdx.y;   // 0..16383
  const int ti = (tile >> 7) << 4;
  const int tj = (tile & 127) << 4;
  const float* Xb = X + (size_t)b * NPTS * SUBSPACE;

  const int mi  = ti + (lane & 15);
  const int kbA = (lane >> 4) << 4;
  const int nj  = tj + (lane & 15);
  const int kbB = (lane >> 4) << 4;

  v8f acc = {};
#pragma unroll
  for (int k0 = 0; k0 < SUBSPACE; k0 += 32) {
    const float* Ai = Xb + (size_t)mi * SUBSPACE + k0;
    const float* Bj = Xb + (size_t)nj * SUBSPACE + k0;   // B(k, n) = X[n][k]
    v16bf af, bfr;
#pragma unroll
    for (int j = 0; j < 8; ++j)  af[j]     = f2bf(Ai[kbA + j]);
#pragma unroll
    for (int j = 0; j < 8; ++j)  af[8 + j] = f2bf(Ai[kbA + 8 + j]);
#pragma unroll
    for (int j = 0; j < 16; ++j) bfr[j]    = f2bf(Bj[kbB + j]);
    acc = __builtin_amdgcn_wmma_f32_16x16x32_bf16(false, af, false, bfr,
                                                  (short)0, acc, false, false);
  }
  const float* sqb = sq + (size_t)b * NPTS;
  const float  sj  = sqb[nj];
  float* D = d2 + (size_t)b * NPTS * NPTS;
  const int rb = ti + ((lane >> 4) << 3);
#pragma unroll
  for (int r = 0; r < 8; ++r) {
    int row = rb + r;
    float v = sqb[row] + sj - 2.0f * acc[r];
    D[(size_t)row * NPTS + nj] = fmaxf(v, 0.0f);
  }
}

// =====================================================================
// Kernel 4: per-row order statistics via value binary search.
// sigma = sqrt(16th smallest incl. self) == 15th smallest without self.
// thr   = 30th smallest incl. self (KNN=30 top-k includes self).
// =====================================================================
__global__ void select_ranks(const float* __restrict__ d2,
                             float* __restrict__ sigma, float* __restrict__ thr) {
  const int tid = threadIdx.x;
  const size_t row = blockIdx.x;
  const float* Drow = d2 + row * (size_t)NPTS;
  __shared__ float sh[256];

  float lo = 3.4e38f, hi = -3.4e38f;
  for (int j = tid; j < NPTS; j += 256) {
    float v = Drow[j];
    lo = fminf(lo, v); hi = fmaxf(hi, v);
  }
  float lo_all = block_red_min(lo, sh);
  float hi_all = block_red_max(hi, sh);

  float res[2];
  const int ks[2] = {16, 30};
  for (int q = 0; q < 2; ++q) {
    float a = lo_all, b2 = hi_all;
    for (int it = 0; it < 40; ++it) {
      float mid = 0.5f * (a + b2);
      float c = 0.f;
      for (int j = tid; j < NPTS; j += 256) c += (Drow[j] <= mid) ? 1.f : 0.f;
      float cnt = block_red_sum(c, sh);
      if (cnt >= (float)ks[q]) b2 = mid; else a = mid;
    }
    res[q] = b2;
  }
  if (tid == 0) {
    sigma[row] = sqrtf(fmaxf(res[0], 1e-12f));
    thr[row]   = res[1];
  }
}

// =====================================================================
// Kernel 5: Kmat (masked Gaussian kernel), in-place over d2; row sums Dg
// =====================================================================
__global__ void kmat_rows(float* __restrict__ d2, const float* __restrict__ sigma,
                          const float* __restrict__ thr, float* __restrict__ Dg) {
  const int tid = threadIdx.x;
  const size_t row = blockIdx.x;
  const int b = (int)(row / NPTS);
  __shared__ float sh[256];
  const float si = sigma[row];
  const float ti = thr[row];
  const float* sigb = sigma + (size_t)b * NPTS;
  const float* thrb = thr   + (size_t)b * NPTS;
  float* Drow = d2 + row * (size_t)NPTS;

  float sum = 0.f;
  for (int j = tid; j < NPTS; j += 256) {
    float dv = Drow[j];
    float kv = __expf(-dv / (si * sigb[j] + 1e-12f));
    bool  m  = (dv <= ti) || (dv <= thrb[j]);     // mask | mask^T
    kv = m ? kv : 0.f;
    Drow[j] = kv;
    sum += kv;
  }
  float tot = block_red_sum(sum, sh);
  if (tid == 0) Dg[row] = fmaxf(tot, 1e-12f);
}

__global__ void compute_dsi(const float* __restrict__ Dg, float* __restrict__ dsi, int n) {
  int i = blockIdx.x * blockDim.x + threadIdx.x;
  if (i >= n) return;
  dsi[i] = rsqrtf(fmaxf(Dg[i], 1e-12f));
}

// Kernel 6: P (in-place over Kmat) and symmetric-normalized S in bf16
__global__ void p_s_rows(float* __restrict__ KP, const float* __restrict__ Dg,
                         const float* __restrict__ dsi, __bf16* __restrict__ Sbf) {
  const size_t row = blockIdx.x;
  const int b = (int)(row / NPTS);
  const float inv = 1.0f / Dg[row];
  const float di  = dsi[row];
  const float* dsib = dsi + (size_t)b * NPTS;
  float*  Krow = KP  + row * (size_t)NPTS;
  __bf16* Srow = Sbf + row * (size_t)NPTS;
  for (int j = threadIdx.x; j < NPTS; j += 256) {
    float kv = Krow[j];
    Srow[j] = f2bf(di * kv * dsib[j]);
    Krow[j] = kv * inv;                           // P
  }
}

// =====================================================================
// Kernel 7: deterministic subspace init
// =====================================================================
__global__ void v_init(float* __restrict__ V, int total) {
  int i = blockIdx.x * blockDim.x + threadIdx.x;
  if (i >= total) return;
  V[i] = __sinf(0.0007f * (float)i + 0.5f);
}

// =====================================================================
// Kernel 8: Y[b] = S[b] @ V[b]   (2048x2048 bf16 @ 2048x64) via WMMA.
// The 32x64 fp32 V k-slab (one contiguous 8 KB block) is async-staged to LDS.
// =====================================================================
__global__ void spmm(const __bf16* __restrict__ S, const float* __restrict__ V,
                     float* __restrict__ Y) {
  __shared__ float Vsh[32 * SUBSPACE];            // 8 KB
  const int lane = threadIdx.x;
  const int tid  = threadIdx.y * 32 + threadIdx.x;
  const int b    = blockIdx.y;
  const int tile = blockIdx.x * blockDim.y + threadIdx.y;   // 0..511 exact
  const int ti = (tile >> 2) << 4;
  const int tj = (tile & 3) << 4;
  const __bf16* Sb = S + (size_t)b * NPTS * NPTS;
  const float*  Vb = V + (size_t)b * NPTS * SUBSPACE;

  const int mi  = ti + (lane & 15);
  const int kbA = (lane >> 4) << 4;
  const int nj  = tj + (lane & 15);
  const int kbB = (lane >> 4) << 4;
  const unsigned lds0 = lds_addr_of(&Vsh[0]);

  v8f acc = {};
  for (int k0 = 0; k0 < NPTS; k0 += 32) {
    // ---- async stage V[k0..k0+31][:] (contiguous 8 KB) into LDS ----
    const char* gsrc = (const char*)(Vb + (size_t)k0 * SUBSPACE);
    async_copy_b128(lds0 + (unsigned)tid * 16u, gsrc + (size_t)tid * 16u);
    async_copy_b128(lds0 + 4096u + (unsigned)tid * 16u, gsrc + 4096u + (size_t)tid * 16u);
    wait_async0();
    __syncthreads();

    const __bf16* Ar = Sb + (size_t)mi * NPTS + k0;
    __builtin_prefetch((const void*)(Ar + 128), 0, 3);
    v16bf af, bfr;
#pragma unroll
    for (int j = 0; j < 8; ++j)  af[j]     = Ar[kbA + j];
#pragma unroll
    for (int j = 0; j < 8; ++j)  af[8 + j] = Ar[kbA + 8 + j];
#pragma unroll
    for (int j = 0; j < 16; ++j)
      bfr[j] = f2bf(Vsh[(kbB + j) * SUBSPACE + nj]);
    acc = __builtin_amdgcn_wmma_f32_16x16x32_bf16(false, af, false, bfr,
                                                  (short)0, acc, false, false);
    __syncthreads();                              // slab reuse barrier
  }
  float* Yb = Y + (size_t)b * NPTS * SUBSPACE;
  const int rb = ti + ((lane >> 4) << 3);
#pragma unroll
  for (int r = 0; r < 8; ++r)
    Yb[(size_t)(rb + r) * SUBSPACE + nj] = acc[r];
}

// =====================================================================
// Kernel 9: modified Gram-Schmidt orthonormalization, Y -> V (per batch)
// =====================================================================
__global__ void mgs(const float* __restrict__ Y, float* __restrict__ V) {
  const int b = blockIdx.x;
  const int tid = threadIdx.x;
  __shared__ float sh[256];
  const float* Yb = Y + (size_t)b * NPTS * SUBSPACE;
  float* Vb = V + (size_t)b * NPTS * SUBSPACE;

  for (int i = tid; i < NPTS * SUBSPACE; i += 256) Vb[i] = Yb[i];
  __syncthreads();

  for (int j = 0; j < SUBSPACE; ++j) {
    for (int i = 0; i < j; ++i) {
      float loc = 0.f;
      for (int r = tid; r < NPTS; r += 256)
        loc += Vb[(size_t)r * SUBSPACE + i] * Vb[(size_t)r * SUBSPACE + j];
      float dot = block_red_sum(loc, sh);
      for (int r = tid; r < NPTS; r += 256)
        Vb[(size_t)r * SUBSPACE + j] -= dot * Vb[(size_t)r * SUBSPACE + i];
      __syncthreads();
    }
    float loc = 0.f;
    for (int r = tid; r < NPTS; r += 256) {
      float v = Vb[(size_t)r * SUBSPACE + j];
      loc += v * v;
    }
    float nrm = block_red_sum(loc, sh);
    float inv = rsqrtf(fmaxf(nrm, 1e-24f));
    for (int r = tid; r < NPTS; r += 256) Vb[(size_t)r * SUBSPACE + j] *= inv;
    __syncthreads();
  }
}

// Kernel 10: Rayleigh quotients lam_j = <V_j, (S V)_j>
__global__ void rayleigh(const float* __restrict__ V, const float* __restrict__ Y,
                         float* __restrict__ lam) {
  const int j = blockIdx.x, b = blockIdx.y, tid = threadIdx.x;
  __shared__ float sh[256];
  const float* Vb = V + (size_t)b * NPTS * SUBSPACE;
  const float* Yb = Y + (size_t)b * NPTS * SUBSPACE;
  float loc = 0.f;
  for (int r = tid; r < NPTS; r += 256)
    loc += Vb[(size_t)r * SUBSPACE + j] * Yb[(size_t)r * SUBSPACE + j];
  float d = block_red_sum(loc, sh);
  if (tid == 0) lam[b * SUBSPACE + j] = d;
}

// Kernel 11: sort eigenvalues descending (tiny, one lane per batch)
__global__ void sortk(const float* __restrict__ lam, int* __restrict__ ord) {
  const int b = blockIdx.x;
  if (threadIdx.x != 0) return;
  int   idx[SUBSPACE];
  float val[SUBSPACE];
  for (int i = 0; i < SUBSPACE; ++i) { idx[i] = i; val[i] = lam[b * SUBSPACE + i]; }
  for (int i = 1; i < SUBSPACE; ++i) {
    float v = val[i]; int id = idx[i]; int k = i - 1;
    while (k >= 0 && val[k] < v) { val[k + 1] = val[k]; idx[k + 1] = idx[k]; --k; }
    val[k + 1] = v; idx[k + 1] = id;
  }
  for (int i = 0; i < SUBSPACE; ++i) ord[b * SUBSPACE + i] = idx[i];
}

// Kernel 12: Psi_t[b][i][t] = evec_{t+1}[i] * dsi[i] * lam_{t+1}
__global__ void psi_build(const float* __restrict__ V, const float* __restrict__ lam,
                          const int* __restrict__ ord, const float* __restrict__ dsi,
                          float* __restrict__ Psi) {
  const size_t row = blockIdx.x;                  // b*N + i
  const int b = (int)(row / NPTS);
  const int i = (int)(row % NPTS);
  const int t = threadIdx.x;
  if (t >= NCL) return;
  const int o = ord[b * SUBSPACE + t + 1];        // skip top eigenvector
  Psi[row * NCL + t] = V[((size_t)b * NPTS + i) * SUBSPACE + o]
                     * dsi[row] * lam[b * SUBSPACE + o];
}

// Kernel 13: diffusion pseudotime distance from ROOT (=0)
__global__ void dpt_kernel(const float* __restrict__ Psi, float* __restrict__ dpt) {
  const size_t row = blockIdx.x;
  const int b = (int)(row / NPTS);
  const int t = threadIdx.x;
  __shared__ float sh[64];
  float loc = 0.f;
  if (t < NCL) {
    float d = Psi[row * NCL + t] - Psi[((size_t)b * NPTS + 0) * NCL + t];
    loc = d * d;
  }
  sh[t] = loc; __syncthreads();
  for (int s = 32; s > 0; s >>= 1) {
    if (t < s) sh[t] += sh[t + s];
    __syncthreads();
  }
  if (t == 0) dpt[row] = sqrtf(sh[0]);
}

// Kernel 14: per-batch tau = max(dpt)/10
__global__ void tau_kernel(const float* __restrict__ dpt, float* __restrict__ tau) {
  const int b = blockIdx.x, tid = threadIdx.x;
  __shared__ float sh[256];
  float m = -1e30f;
  for (int i = tid; i < NPTS; i += 256) m = fmaxf(m, dpt[(size_t)b * NPTS + i]);
  float mx = block_red_max(m, sh);
  if (tid == 0) tau[b] = fmaxf(mx, 1e-6f) * 0.1f;
}

// Kernel 15: A = normalize_rows(P * soft(dt))
__global__ void final_a(const float* __restrict__ P, const float* __restrict__ dpt,
                        const float* __restrict__ tau, float* __restrict__ A) {
  const size_t row = blockIdx.x;
  const int b = (int)(row / NPTS);
  const int tid = threadIdx.x;
  __shared__ float sh[256];
  const float* Prow = P + row * (size_t)NPTS;
  float* Arow = A + row * (size_t)NPTS;
  const float* dptb = dpt + (size_t)b * NPTS;
  const float di = dpt[row];
  const float itau = 1.0f / tau[b];
  float loc = 0.f;
  for (int j = tid; j < NPTS; j += 256) {
    float dt = di - dptb[j];
    float s = 0.f;
    if (dt > 0.f) { float z = dt * itau; s = __expf(-z * z); }
    float a = Prow[j] * s;
    Arow[j] = a;
    loc += a;
  }
  float sum = block_red_sum(loc, sh);
  float inv = 1.0f / fmaxf(sum, 1e-12f);
  for (int j = tid; j < NPTS; j += 256) Arow[j] *= inv;
}

// =====================================================================
extern "C" void kernel_launch(void* const* d_in, const int* in_sizes, int n_in,
                              void* d_out, int out_size, void* d_ws, size_t ws_size,
                              hipStream_t stream) {
  (void)in_sizes; (void)n_in; (void)out_size; (void)ws_size;
  const float* emb = (const float*)d_in[0];
  const float* W1  = (const float*)d_in[1];
  const float* b1  = (const float*)d_in[2];
  const float* W2  = (const float*)d_in[3];
  const float* b2  = (const float*)d_in[4];
  const float* W3  = (const float*)d_in[5];
  const float* b3  = (const float*)d_in[6];

  char* ws = (char*)d_ws;
  size_t off = 0;
  auto carve = [&](size_t bytes) -> void* {
    void* p = (void*)(ws + off);
    off += (bytes + 255) & ~(size_t)255;
    return p;
  };

  float*  H1  = (float*)carve((size_t)ROWS * 128 * sizeof(float));
  float*  H2  = (float*)carve((size_t)ROWS * SUBSPACE * sizeof(float));
  float*  X   = (float*)carve((size_t)ROWS * SUBSPACE * sizeof(float));
  float*  sq  = (float*)carve((size_t)ROWS * sizeof(float));
  float*  sig = (float*)carve((size_t)ROWS * sizeof(float));
  float*  thr = (float*)carve((size_t)ROWS * sizeof(float));
  float*  Dg  = (float*)carve((size_t)ROWS * sizeof(float));
  float*  dsi = (float*)carve((size_t)ROWS * sizeof(float));
  float*  KP  = (float*)carve((size_t)BSZ * NPTS * NPTS * sizeof(float));   // d2 -> Kmat -> P
  __bf16* Sbf = (__bf16*)carve((size_t)BSZ * NPTS * NPTS * sizeof(__bf16));
  float*  V   = (float*)carve((size_t)BSZ * NPTS * SUBSPACE * sizeof(float));
  float*  Y   = (float*)carve((size_t)BSZ * NPTS * SUBSPACE * sizeof(float));
  float*  lam = (float*)carve((size_t)BSZ * SUBSPACE * sizeof(float));
  int*    ord = (int*)  carve((size_t)BSZ * SUBSPACE * sizeof(int));
  float*  Psi = (float*)carve((size_t)ROWS * NCL * sizeof(float));
  float*  dpt = (float*)carve((size_t)ROWS * sizeof(float));
  float*  tau = (float*)carve((size_t)BSZ * sizeof(float));

  const dim3 wblock(32, 8);   // 8 waves of 32 (wave32), one 16x16 tile per wave

  // MLP (bf16 WMMA, fp32 accumulate)
  mlp_layer<<<dim3((ROWS / 16) * (128 / 16) / 8), wblock, 0, stream>>>(
      emb, W1, b1, H1, ROWS, DIM, 128, 128, 1);
  mlp_layer<<<dim3((ROWS / 16) * (64 / 16) / 8), wblock, 0, stream>>>(
      H1, W2, b2, H2, ROWS, 128, 64, 64, 1);
  mlp_layer<<<dim3((ROWS / 16) * (64 / 16) / 8), wblock, 0, stream>>>(
      H2, W3, b3, X, ROWS, 64, NCL, 64, 0);

  // pairwise distances
  row_sqnorm<<<dim3(ROWS / 256), dim3(256), 0, stream>>>(X, sq, ROWS);
  gram_d2<<<dim3((128 * 128) / 8, BSZ), wblock, 0, stream>>>(X, sq, KP);

  // order statistics, kernel matrix, normalizations
  select_ranks<<<dim3(ROWS), dim3(256), 0, stream>>>(KP, sig, thr);
  kmat_rows<<<dim3(ROWS), dim3(256), 0, stream>>>(KP, sig, thr, Dg);
  compute_dsi<<<dim3(ROWS / 256), dim3(256), 0, stream>>>(Dg, dsi, ROWS);
  p_s_rows<<<dim3(ROWS), dim3(256), 0, stream>>>(KP, Dg, dsi, Sbf);

  // spectral embedding via subspace iteration (WMMA GEMMs)
  v_init<<<dim3((BSZ * NPTS * SUBSPACE) / 256), dim3(256), 0, stream>>>(
      V, BSZ * NPTS * SUBSPACE);
  for (int it = 0; it < SUB_ITERS; ++it) {
    spmm<<<dim3(512 / 8, BSZ), wblock, 0, stream>>>(Sbf, V, Y);
    mgs<<<dim3(BSZ), dim3(256), 0, stream>>>(Y, V);
  }
  spmm<<<dim3(512 / 8, BSZ), wblock, 0, stream>>>(Sbf, V, Y);
  rayleigh<<<dim3(SUBSPACE, BSZ), dim3(256), 0, stream>>>(V, Y, lam);
  sortk<<<dim3(BSZ), dim3(32), 0, stream>>>(lam, ord);

  // pseudotime + directed transition matrix
  psi_build<<<dim3(ROWS), dim3(64), 0, stream>>>(V, lam, ord, dsi, Psi);
  dpt_kernel<<<dim3(ROWS), dim3(64), 0, stream>>>(Psi, dpt);
  tau_kernel<<<dim3(BSZ), dim3(256), 0, stream>>>(dpt, tau);
  final_a<<<dim3(ROWS), dim3(256), 0, stream>>>(KP, dpt, tau, (float*)d_out);
}
